// BottleNeckAudioVideoRMAttnT_74655121539158
// MI455X (gfx1250) — matrix operand order repaired and checked by reading, not compiled, not run on hardware
//
#include <hip/hip_runtime.h>
#include <hip/hip_bf16.h>
#include <math.h>

// ---------------------------------------------------------------------------
// Problem constants (from reference)
// ---------------------------------------------------------------------------
#define BB_   2
#define TT_   2048
#define CC_   512
#define AA_   128
#define VV_   512
#define OUTC_ 256
#define NH_   8
#define HS_   64
#define MROWS (BB_ * TT_)   // 4096 token rows

typedef float        v8f   __attribute__((ext_vector_type(8)));
typedef __bf16       v16bf __attribute__((ext_vector_type(16)));
typedef __bf16       v8bf  __attribute__((ext_vector_type(8)));
typedef unsigned int u32x4 __attribute__((ext_vector_type(4)));
typedef int          i32x4 __attribute__((ext_vector_type(4)));
typedef int          i32x8 __attribute__((ext_vector_type(8)));

#ifndef __has_builtin
#define __has_builtin(x) 0
#endif
#if __has_builtin(__builtin_amdgcn_tensor_load_to_lds) && __has_builtin(__builtin_amdgcn_s_wait_tensorcnt)
#define USE_TDM 1
#else
#define USE_TDM 0
#endif

__device__ __forceinline__ float gelu_f(float x) {
    return 0.5f * x * (1.0f + erff(x * 0.70710678118654752f));
}

enum { EPI_NONE = 0, EPI_TANH = 1, EPI_GELU = 2, EPI_RESMASK = 3, EPI_RES = 4 };

// ---------------------------------------------------------------------------
// fp32 -> bf16 weight pre-conversion (once per launch; weights reused 32x/GEMM)
// ---------------------------------------------------------------------------
__global__ void cvt_bf16_kernel(const float* __restrict__ src,
                                __bf16* __restrict__ dst, int n)
{
    const int i = blockIdx.x * blockDim.x + threadIdx.x;
    if (i < n) dst[i] = (__bf16)src[i];
}

// ---------------------------------------------------------------------------
// WMMA bf16 GEMM:  C[M,N] = epi( A[M,K](fp32) * Wbf[N,K]^T(bf16) + bias[N] )
//   ATRANS: A element (row,k) read from channel-major (B,K,T) layout
//   CTRANS: store to (B,N,T) layout (output head)
// Tile 128x64, 256 threads = 8 waves, each wave 32x32 = 2x2 WMMA tiles.
// B tile moved global->LDS by the Tensor Data Mover (pad 64B rows to 80B).
// ---------------------------------------------------------------------------
#define GBM 128
#define GBN 64
#define GBK 32
#define LDS_STRIDE (GBK + 8)   // 40 bf16 = 80B row stride (16B aligned)

template<int EPI, bool ATRANS, bool CTRANS>
__global__ __launch_bounds__(256, 2)
void gemm_bf16_wmma_kernel(const float* __restrict__ A,
                           const __bf16* __restrict__ Wbf,
                           const float* __restrict__ bias,
                           float* __restrict__ Cout,
                           const float* __restrict__ resid,
                           const unsigned char* __restrict__ maskRow,
                           int Mdim, int Ndim, int Kdim, int Td)
{
    __shared__ alignas(16) __bf16 As[GBM][LDS_STRIDE];
    __shared__ alignas(16) __bf16 Bs[GBN][LDS_STRIDE];

    const int tid  = threadIdx.x;
    const int lane = tid & 31;
    const int wave = tid >> 5;
    const int wm   = wave & 3;   // 4 waves along M (32 rows each)
    const int wn   = wave >> 2;  // 2 waves along N (32 cols each)

    const int tileM = blockIdx.x * GBM;
    const int tileN = blockIdx.y * GBN;

    v8f acc[2][2] = {};

    const int arow  = tid >> 1;  // 0..127
    const int ahalf = tid & 1;   // 16-col half
    const int brow  = tid >> 2;  // 0..63  (fallback path)
    const int bqrt  = tid & 3;

    // fragment addressing (CDNA5 wave32 WMMA layouts)
    const int mA  = lane & 15;
    const int c0A = (lane >> 4) * 8;    // A: lane group K-half
    const int nB  = lane & 15;
    const int kB  = (lane >> 4) * 8;    // B: lane group K-half

#if USE_TDM
    // constant parts of the D# descriptor for the B tile
    const unsigned int ldsB = (unsigned int)(uintptr_t)&Bs[0][0];
    // group1: data_size=2B, pad_enable, pad_interval=16 DW (64B), pad_amount=4 DW (16B)
    i32x8 g1;
    g1[0] = (1 << 16) | (1 << 20) | (3 << 22) | (3 << 25);
    g1[1] = (Kdim & 0xFFFF) << 16;                      // tensor_dim0 lo16
    g1[2] = ((Kdim >> 16) & 0xFFFF) | ((Ndim & 0xFFFF) << 16);
    g1[3] = ((Ndim >> 16) & 0xFFFF) | (GBK << 16);      // tile_dim0 = 32
    g1[4] = GBN;                                        // tile_dim1 = 64, tile_dim2 = 0
    g1[5] = Kdim;                                       // tensor_dim0_stride
    g1[6] = 0;
    g1[7] = 0;
    const i32x4 gz = {0, 0, 0, 0};
#if __has_include(<hip/amd_detail/amd_gfx1250_TDM.h>)
    const i32x8 gz8 = {0, 0, 0, 0, 0, 0, 0, 0};
#endif
#endif

    for (int k0 = 0; k0 < Kdim; k0 += GBK) {
        // ---- B tile: TDM async copy (bf16, already converted) ----
#if USE_TDM
        if (wave == 0) {
            const unsigned long long gaddr =
                (unsigned long long)(uintptr_t)(Wbf + (size_t)tileN * Kdim + k0);
            u32x4 g0;
            g0[0] = 1u;                                  // count=1, user mode
            g0[1] = ldsB;                                // LDS byte address
            g0[2] = (unsigned int)(gaddr & 0xFFFFFFFFu);
            g0[3] = (unsigned int)(gaddr >> 32) | (2u << 30);  // type=2 (image)
#if __has_include(<hip/amd_detail/amd_gfx1250_TDM.h>)
            __builtin_amdgcn_tensor_load_to_lds(g0, g1, gz, gz, gz8, 0);
#else
            __builtin_amdgcn_tensor_load_to_lds(g0, g1, gz, gz, 0);
#endif
        }
#else
        {   // fallback: one b128 per thread (64 rows x 32 bf16)
            const __bf16* src = Wbf + (size_t)(tileN + brow) * Kdim + k0 + bqrt * 8;
            *(v8bf*)&Bs[brow][bqrt * 8] = *(const v8bf*)src;
        }
#endif
        // ---- A tile: fp32 -> bf16 staging ----
        {
            const int grow = tileM + arow;
            if (!ATRANS) {
                const float* src = A + (size_t)grow * Kdim + k0 + ahalf * 16;
                if (k0 + GBK < Kdim) __builtin_prefetch(src + GBK, 0, 0);
                #pragma unroll
                for (int u = 0; u < 4; ++u) {
                    const float4 f = ((const float4*)src)[u];
                    As[arow][ahalf * 16 + u * 4 + 0] = (__bf16)f.x;
                    As[arow][ahalf * 16 + u * 4 + 1] = (__bf16)f.y;
                    As[arow][ahalf * 16 + u * 4 + 2] = (__bf16)f.z;
                    As[arow][ahalf * 16 + u * 4 + 3] = (__bf16)f.w;
                }
            } else {
                const int bidx = grow / Td;
                const int tpos = grow % Td;
                const float* src = A + ((size_t)bidx * Kdim + k0 + ahalf * 16) * Td + tpos;
                #pragma unroll
                for (int i = 0; i < 16; ++i)
                    As[arow][ahalf * 16 + i] = (__bf16)src[(size_t)i * Td];
            }
        }
#if USE_TDM
        if (wave == 0) __builtin_amdgcn_s_wait_tensorcnt(0);
#endif
        __syncthreads();

        // ---- build fragments from LDS ----
        v16bf afrag[2], bfrag[2];
        #pragma unroll
        for (int i = 0; i < 2; ++i) {
            const __bf16* pa = &As[wm * 32 + i * 16 + mA][c0A];
            v8bf lo = *(const v8bf*)pa;          // K = c0A .. c0A+7
            v8bf hi = *(const v8bf*)(pa + 16);   // K = c0A+16 .. c0A+23
            #pragma unroll
            for (int e = 0; e < 8; ++e) { afrag[i][e] = lo[e]; afrag[i][e + 8] = hi[e]; }
        }
        #pragma unroll
        for (int j = 0; j < 2; ++j) {
            const __bf16* pb = &Bs[wn * 32 + j * 16 + nB][kB];
            v8bf lo = *(const v8bf*)pb;          // K = kB .. kB+7
            v8bf hi = *(const v8bf*)(pb + 16);   // K = kB+16 .. kB+23
            #pragma unroll
            for (int e = 0; e < 8; ++e) { bfrag[j][e] = lo[e]; bfrag[j][e + 8] = hi[e]; }
        }

        // ---- 2x2 WMMA ----
        #pragma unroll
        for (int i = 0; i < 2; ++i)
            #pragma unroll
            for (int j = 0; j < 2; ++j)
                acc[i][j] = __builtin_amdgcn_wmma_f32_16x16x32_bf16(
                    false, afrag[i], false, bfrag[j],
                    (short)0, acc[i][j], false, false);
        __syncthreads();
    }

    // ---- epilogue ----
    const int mbase = tileM + wm * 32;
    const int nbase = tileN + wn * 32;
    #pragma unroll
    for (int i = 0; i < 2; ++i) {
        #pragma unroll
        for (int j = 0; j < 2; ++j) {
            #pragma unroll
            for (int r = 0; r < 8; ++r) {
                const int grow = mbase + i * 16 + (lane >> 4) * 8 + r; // D: M = r + 8*(lane/16)
                const int gcol = nbase + j * 16 + (lane & 15);         // D: N = lane%16
                float val = acc[i][j][r] + bias[gcol];
                if (EPI == EPI_TANH) val = tanhf(val);
                if (EPI == EPI_GELU) val = gelu_f(val);
                if (EPI == EPI_RESMASK) {
                    const float m = maskRow[grow] ? 1.0f : 0.0f;
                    val = (resid[(size_t)grow * Ndim + gcol] + val) * m;
                }
                if (EPI == EPI_RES) {
                    const float m = maskRow[grow] ? 1.0f : 0.0f;
                    val = resid[(size_t)grow * Ndim + gcol] * m + val;
                }
                if (!CTRANS) {
                    Cout[(size_t)grow * Ndim + gcol] = val;
                } else {
                    const int bidx = grow / Td;
                    const int tpos = grow % Td;
                    Cout[((size_t)bidx * Ndim + gcol) * Td + tpos] = val;
                }
            }
        }
    }
}

// ---------------------------------------------------------------------------
// elementwise combine: x = tanh(ev+em) + tanh(ea+em), em = ev+ea
// ---------------------------------------------------------------------------
__global__ void combine_kernel(const float* __restrict__ ev,
                               const float* __restrict__ ea,
                               float* __restrict__ x, int n)
{
    int i = blockIdx.x * blockDim.x + threadIdx.x;
    if (i < n) {
        float a = ea[i], v = ev[i];
        float em = a + v;
        x[i] = tanhf(v + em) + tanhf(a + em);
    }
}

// ---------------------------------------------------------------------------
// block LN helper over 512 channels (256 threads, 2 channels/thread)
// ---------------------------------------------------------------------------
__device__ __forceinline__ void block_ln_512(float y0, float y1, int c0, int c1,
                                             const float* __restrict__ g,
                                             const float* __restrict__ b,
                                             float* __restrict__ outRow,
                                             float scale, float* red, int tid)
{
    red[tid] = y0 + y1;
    __syncthreads();
    for (int s = 128; s > 0; s >>= 1) { if (tid < s) red[tid] += red[tid + s]; __syncthreads(); }
    const float mu = red[0] * (1.0f / 512.0f);
    __syncthreads();
    const float r0 = y0 - mu, r1 = y1 - mu;
    red[tid] = r0 * r0 + r1 * r1;
    __syncthreads();
    for (int s = 128; s > 0; s >>= 1) { if (tid < s) red[tid] += red[tid + s]; __syncthreads(); }
    const float inv = rsqrtf(red[0] * (1.0f / 512.0f) + 1e-5f);
    __syncthreads();
    outRow[c0] = (r0 * inv * g[c0] + b[c0]) * scale;
    outRow[c1] = (r1 * inv * g[c1] + b[c1]) * scale;
}

// channel LayerNorm per (b,t) row; optional post-mask
__global__ __launch_bounds__(256)
void cln_kernel(const float* __restrict__ X, const float* __restrict__ g,
                const float* __restrict__ b, const unsigned char* __restrict__ mask,
                int useMask, float* __restrict__ Y)
{
    __shared__ float red[256];
    const int row = blockIdx.x, tid = threadIdx.x;
    const float* x = X + (size_t)row * CC_;
    const float v0 = x[tid], v1 = x[tid + 256];
    const float m = useMask ? (mask[row] ? 1.0f : 0.0f) : 1.0f;
    block_ln_512(v0, v1, tid, tid + 256, g, b, Y + (size_t)row * CC_, m, red, tid);
}

// depthwise conv3 (along t, zero-pad per batch) * mask, then LN — for q,k,v
__global__ __launch_bounds__(256)
void dwcln_kernel(const float* __restrict__ xn,
                  const float* __restrict__ qw, const float* __restrict__ qg, const float* __restrict__ qb,
                  const float* __restrict__ kw, const float* __restrict__ kg, const float* __restrict__ kb,
                  const float* __restrict__ vw, const float* __restrict__ vg, const float* __restrict__ vb,
                  const unsigned char* __restrict__ mask,
                  float* __restrict__ qo, float* __restrict__ ko, float* __restrict__ vo)
{
    __shared__ float red[256];
    const int row = blockIdx.x, tid = threadIdx.x;
    const int t = row % TT_;
    const float m = mask[row] ? 1.0f : 0.0f;
    const float* x0 = xn + (size_t)row * CC_;
    const int c0 = tid, c1 = tid + 256;

    const float a0 = x0[c0],  a1 = x0[c1];
    const float p0 = (t > 0)       ? x0[c0 - CC_] : 0.0f;
    const float p1 = (t > 0)       ? x0[c1 - CC_] : 0.0f;
    const float n0 = (t < TT_ - 1) ? x0[c0 + CC_] : 0.0f;
    const float n1 = (t < TT_ - 1) ? x0[c1 + CC_] : 0.0f;

    float y0, y1;
    y0 = (qw[c0*3] * p0 + qw[c0*3+1] * a0 + qw[c0*3+2] * n0) * m;
    y1 = (qw[c1*3] * p1 + qw[c1*3+1] * a1 + qw[c1*3+2] * n1) * m;
    block_ln_512(y0, y1, c0, c1, qg, qb, qo + (size_t)row * CC_, 1.0f, red, tid);

    y0 = (kw[c0*3] * p0 + kw[c0*3+1] * a0 + kw[c0*3+2] * n0) * m;
    y1 = (kw[c1*3] * p1 + kw[c1*3+1] * a1 + kw[c1*3+2] * n1) * m;
    block_ln_512(y0, y1, c0, c1, kg, kb, ko + (size_t)row * CC_, 1.0f, red, tid);

    y0 = (vw[c0*3] * p0 + vw[c0*3+1] * a0 + vw[c0*3+2] * n0) * m;
    y1 = (vw[c1*3] * p1 + vw[c1*3+1] * a1 + vw[c1*3+2] * n1) * m;
    block_ln_512(y0, y1, c0, c1, vg, vb, vo + (size_t)row * CC_, 1.0f, red, tid);
}

// ---------------------------------------------------------------------------
// local windowed attention (window +-9), one thread per (b,h,t)
// ---------------------------------------------------------------------------
__global__ __launch_bounds__(256)
void local_attn_kernel(const float* __restrict__ q, const float* __restrict__ k,
                       const float* __restrict__ v, const unsigned char* __restrict__ mask,
                       float* __restrict__ o)
{
    const int gidx = blockIdx.x * 256 + threadIdx.x;   // B*NH*T threads
    const int b   = gidx / (NH_ * TT_);
    const int rem = gidx - b * NH_ * TT_;
    const int h   = rem / TT_;
    const int t   = rem - h * TT_;

    const float4* q4 = (const float4*)(q + (size_t)(b * TT_ + t) * CC_ + h * HS_);
    float att[19];
    float mx = -3.0e38f;
    for (int w = 0; w < 19; ++w) {
        const int tt = t + w - 9;
        if (tt < 0 || tt >= TT_) { att[w] = -3.0e38f; continue; }
        const float4* k4 = (const float4*)(k + (size_t)(b * TT_ + tt) * CC_ + h * HS_);
        float s = 0.0f;
        #pragma unroll
        for (int d = 0; d < 16; ++d) {
            const float4 qa = q4[d], ka = k4[d];
            s += qa.x * ka.x + qa.y * ka.y + qa.z * ka.z + qa.w * ka.w;
        }
        s *= 0.125f;                                  // 1/sqrt(64)
        if (!mask[b * TT_ + tt]) s -= 10000.0f;
        att[w] = s;
        mx = fmaxf(mx, s);
    }
    float denom = 0.0f;
    for (int w = 0; w < 19; ++w) {
        if (att[w] > -1.0e38f) { att[w] = expf(att[w] - mx); denom += att[w]; }
        else                     att[w] = 0.0f;
    }
    const float inv = (mask[b * TT_ + t] ? 1.0f : 0.0f) / denom;
    for (int w = 0; w < 19; ++w) att[w] *= inv;

    float4* o4 = (float4*)(o + (size_t)(b * TT_ + t) * CC_ + h * HS_);
    #pragma unroll 4
    for (int d = 0; d < 16; ++d) {
        float4 acc = {0.0f, 0.0f, 0.0f, 0.0f};
        for (int w = 0; w < 19; ++w) {
            if (att[w] == 0.0f) continue;
            const int tt = t + w - 9;
            const float4 vv = ((const float4*)(v + (size_t)(b * TT_ + tt) * CC_ + h * HS_))[d];
            acc.x += att[w] * vv.x; acc.y += att[w] * vv.y;
            acc.z += att[w] * vv.z; acc.w += att[w] * vv.w;
        }
        o4[d] = acc;
    }
}

// ---------------------------------------------------------------------------
// launch
// ---------------------------------------------------------------------------
extern "C" void kernel_launch(void* const* d_in, const int* in_sizes, int n_in,
                              void* d_out, int out_size, void* d_ws, size_t ws_size,
                              hipStream_t stream)
{
    (void)in_sizes; (void)n_in; (void)out_size; (void)ws_size;
    const float* emb_v = (const float*)d_in[0];
    const float* emb_a = (const float*)d_in[1];
    const float* w_v1  = (const float*)d_in[2];
    const float* b_v1  = (const float*)d_in[3];
    const float* w_a1  = (const float*)d_in[4];
    const float* b_a1  = (const float*)d_in[5];
    const float* w_v2  = (const float*)d_in[6];
    const float* b_v2  = (const float*)d_in[7];
    const float* w_a2  = (const float*)d_in[8];
    const float* b_a2  = (const float*)d_in[9];
    const float* ln1_g = (const float*)d_in[10];
    const float* ln1_b = (const float*)d_in[11];
    const float* qconv = (const float*)d_in[12];
    const float* qng   = (const float*)d_in[13];
    const float* qnb   = (const float*)d_in[14];
    const float* kconv = (const float*)d_in[15];
    const float* kng   = (const float*)d_in[16];
    const float* knb   = (const float*)d_in[17];
    const float* vconv = (const float*)d_in[18];
    const float* vng   = (const float*)d_in[19];
    const float* vnb   = (const float*)d_in[20];
    const float* wq    = (const float*)d_in[21];
    const float* bq    = (const float*)d_in[22];
    const float* wk    = (const float*)d_in[23];
    const float* bk    = (const float*)d_in[24];
    const float* wv    = (const float*)d_in[25];
    const float* bv    = (const float*)d_in[26];
    const float* wp    = (const float*)d_in[27];
    const float* bp    = (const float*)d_in[28];
    const float* ln2_g = (const float*)d_in[29];
    const float* ln2_b = (const float*)d_in[30];
    const float* mw1   = (const float*)d_in[31];
    const float* mb1   = (const float*)d_in[32];
    const float* mw2   = (const float*)d_in[33];
    const float* mb2   = (const float*)d_in[34];
    const float* w_out = (const float*)d_in[35];
    const float* b_out = (const float*)d_in[36];
    const unsigned char* mask = (const unsigned char*)d_in[37];

    // workspace: 6 x 8MB fp32 slots, 32MB hidden, then bf16 weight pool
    const size_t SLOT = (size_t)MROWS * CC_ * sizeof(float);
    char* ws = (char*)d_ws;
    float* S0 = (float*)(ws + 0 * SLOT);
    float* S1 = (float*)(ws + 1 * SLOT);
    float* S2 = (float*)(ws + 2 * SLOT);
    float* S3 = (float*)(ws + 3 * SLOT);
    float* S4 = (float*)(ws + 4 * SLOT);
    float* S5 = (float*)(ws + 5 * SLOT);
    float* Hb = (float*)(ws + 6 * SLOT);                       // 4096 x 2048 fp32
    char*  wpool = ws + 6 * SLOT + (size_t)MROWS * 2048 * 4;   // bf16 weights

    const dim3 blk(256);

    // ---- pre-convert all GEMM weights to bf16 ----
    size_t off = 0;
    auto wslot = [&](int nelem) {
        __bf16* p = (__bf16*)(wpool + off);
        off += ((size_t)nelem * 2 + 255) & ~(size_t)255;
        return p;
    };
    __bf16* w_a1b = wslot(CC_ * AA_);
    __bf16* w_v1b = wslot(CC_ * VV_);
    __bf16* w_a2b = wslot(CC_ * CC_);
    __bf16* w_v2b = wslot(CC_ * CC_);
    __bf16* wqb   = wslot(CC_ * CC_);
    __bf16* wkb   = wslot(CC_ * CC_);
    __bf16* wvb   = wslot(CC_ * CC_);
    __bf16* wpb   = wslot(CC_ * CC_);
    __bf16* mw1b  = wslot(2048 * CC_);
    __bf16* mw2b  = wslot(CC_ * 2048);
    __bf16* woutb = wslot(OUTC_ * CC_);
    cvt_bf16_kernel<<<(CC_ * AA_) / 256, blk, 0, stream>>>(w_a1, w_a1b, CC_ * AA_);
    cvt_bf16_kernel<<<(CC_ * VV_) / 256, blk, 0, stream>>>(w_v1, w_v1b, CC_ * VV_);
    cvt_bf16_kernel<<<(CC_ * CC_) / 256, blk, 0, stream>>>(w_a2, w_a2b, CC_ * CC_);
    cvt_bf16_kernel<<<(CC_ * CC_) / 256, blk, 0, stream>>>(w_v2, w_v2b, CC_ * CC_);
    cvt_bf16_kernel<<<(CC_ * CC_) / 256, blk, 0, stream>>>(wq, wqb, CC_ * CC_);
    cvt_bf16_kernel<<<(CC_ * CC_) / 256, blk, 0, stream>>>(wk, wkb, CC_ * CC_);
    cvt_bf16_kernel<<<(CC_ * CC_) / 256, blk, 0, stream>>>(wv, wvb, CC_ * CC_);
    cvt_bf16_kernel<<<(CC_ * CC_) / 256, blk, 0, stream>>>(wp, wpb, CC_ * CC_);
    cvt_bf16_kernel<<<(2048 * CC_) / 256, blk, 0, stream>>>(mw1, mw1b, 2048 * CC_);
    cvt_bf16_kernel<<<(CC_ * 2048) / 256, blk, 0, stream>>>(mw2, mw2b, CC_ * 2048);
    cvt_bf16_kernel<<<(OUTC_ * CC_) / 256, blk, 0, stream>>>(w_out, woutb, OUTC_ * CC_);

    const dim3 g512(MROWS / GBM, CC_ / GBN);     // (32, 8)
    const dim3 g2048(MROWS / GBM, 2048 / GBN);   // (32, 32)
    const dim3 g256(MROWS / GBM, OUTC_ / GBN);   // (32, 4)

    // audio MLP
    gemm_bf16_wmma_kernel<EPI_TANH, true,  false><<<g512, blk, 0, stream>>>(
        emb_a, w_a1b, b_a1, S0, nullptr, nullptr, MROWS, CC_, AA_, TT_);
    gemm_bf16_wmma_kernel<EPI_NONE, false, false><<<g512, blk, 0, stream>>>(
        S0, w_a2b, b_a2, S1, nullptr, nullptr, MROWS, CC_, CC_, TT_);
    // video MLP
    gemm_bf16_wmma_kernel<EPI_TANH, true,  false><<<g512, blk, 0, stream>>>(
        emb_v, w_v1b, b_v1, S0, nullptr, nullptr, MROWS, CC_, VV_, TT_);
    gemm_bf16_wmma_kernel<EPI_NONE, false, false><<<g512, blk, 0, stream>>>(
        S0, w_v2b, b_v2, S2, nullptr, nullptr, MROWS, CC_, CC_, TT_);
    // x = tanh(ev+em)+tanh(ea+em) -> S3
    combine_kernel<<<(MROWS * CC_) / 256, blk, 0, stream>>>(S2, S1, S3, MROWS * CC_);
    // xn = LN(x) -> S4
    cln_kernel<<<MROWS, blk, 0, stream>>>(S3, ln1_g, ln1_b, nullptr, 0, S4);
    // q/k/v inputs: conv3*mask + LN -> S0,S1,S2
    dwcln_kernel<<<MROWS, blk, 0, stream>>>(S4, qconv, qng, qnb, kconv, kng, knb,
                                            vconv, vng, vnb, mask, S0, S1, S2);
    // projections
    gemm_bf16_wmma_kernel<EPI_NONE, false, false><<<g512, blk, 0, stream>>>(
        S0, wqb, bq, S4, nullptr, nullptr, MROWS, CC_, CC_, TT_);   // q -> S4
    gemm_bf16_wmma_kernel<EPI_NONE, false, false><<<g512, blk, 0, stream>>>(
        S1, wkb, bk, S0, nullptr, nullptr, MROWS, CC_, CC_, TT_);   // k -> S0
    gemm_bf16_wmma_kernel<EPI_NONE, false, false><<<g512, blk, 0, stream>>>(
        S2, wvb, bv, S1, nullptr, nullptr, MROWS, CC_, CC_, TT_);   // v -> S1
    // local attention -> S2
    local_attn_kernel<<<(BB_ * NH_ * TT_) / 256, blk, 0, stream>>>(S4, S0, S1, mask, S2);
    // out = (x + o@wp^T + bp) * mask -> S5
    gemm_bf16_wmma_kernel<EPI_RESMASK, false, false><<<g512, blk, 0, stream>>>(
        S2, wpb, bp, S5, S3, mask, MROWS, CC_, CC_, TT_);
    // LN2 * mask -> S0
    cln_kernel<<<MROWS, blk, 0, stream>>>(S5, ln2_g, ln2_b, mask, 1, S0);
    // MLP
    gemm_bf16_wmma_kernel<EPI_GELU, false, false><<<g2048, blk, 0, stream>>>(
        S0, mw1b, mb1, Hb, nullptr, nullptr, MROWS, 2048, CC_, TT_);
    gemm_bf16_wmma_kernel<EPI_RES, false, false><<<g512, blk, 0, stream>>>(
        Hb, mw2b, mb2, S1, S5, mask, MROWS, CC_, 2048, TT_);
    // head -> d_out (B,OUT,T)
    gemm_bf16_wmma_kernel<EPI_GELU, false, true><<<g256, blk, 0, stream>>>(
        S1, woutb, b_out, (float*)d_out, nullptr, nullptr, MROWS, OUTC_, CC_, TT_);
}
